// SpatialAttentionLayer_15582141350418
// MI455X (gfx1250) — matrix operand order repaired and checked by reading, not compiled, not run on hardware
//
#include <hip/hip_runtime.h>

typedef __attribute__((ext_vector_type(16))) __bf16 v16bf;
typedef __attribute__((ext_vector_type(8)))  __bf16 v8bf;
typedef __attribute__((ext_vector_type(8)))  float  v8f;

#define BATCH 16
#define CCH   64
#define STOK  4096

// Q scale: 1/sqrt(C) * log2(e), so softmax can use raw v_exp_f32 (base-2).
#define QSCALE 0.18033688011112042f

// ---------------------------------------------------------------------------
// DPP helpers: reduce over the 16-lane row group (wave32 halves).
// ---------------------------------------------------------------------------
template <int CTRL>
__device__ __forceinline__ float dpp_mov_f32(float x) {
    int v = __builtin_amdgcn_update_dpp(__float_as_int(x), __float_as_int(x),
                                        CTRL, 0xF, 0xF, true);
    return __int_as_float(v);
}
__device__ __forceinline__ float row_max16(float x) {
    x = fmaxf(x, dpp_mov_f32<0xB1>(x));    // quad_perm(1,0,3,2)
    x = fmaxf(x, dpp_mov_f32<0x4E>(x));    // quad_perm(2,3,0,1)
    x = fmaxf(x, dpp_mov_f32<0x141>(x));   // row_half_mirror
    x = fmaxf(x, dpp_mov_f32<0x140>(x));   // row_mirror
    return x;
}
__device__ __forceinline__ float row_sum16(float x) {
    x += dpp_mov_f32<0xB1>(x);
    x += dpp_mov_f32<0x4E>(x);
    x += dpp_mov_f32<0x141>(x);
    x += dpp_mov_f32<0x140>(x);
    return x;
}

// contiguous 16 bf16 -> v16bf (two 16B loads)
__device__ __forceinline__ v16bf ld_frag16(const __bf16* p) {
    v8bf lo = *(const v8bf*)(p);
    v8bf hi = *(const v8bf*)(p + 8);
    return __builtin_shufflevector(lo, hi, 0,1,2,3,4,5,6,7,8,9,10,11,12,13,14,15);
}
// A-fragment style: 8 bf16 at p, 8 bf16 at p+16
__device__ __forceinline__ v16bf ld_frag8_8(const __bf16* p) {
    v8bf lo = *(const v8bf*)(p);
    v8bf hi = *(const v8bf*)(p + 16);
    return __builtin_shufflevector(lo, hi, 0,1,2,3,4,5,6,7,8,9,10,11,12,13,14,15);
}

// ---------------------------------------------------------------------------
// Kernel 1: QKV projections (1x1 convs). x is [B,C,S] (S = H*W).
// ---------------------------------------------------------------------------
__global__ __launch_bounds__(256) void qkv_kernel(
    const float* __restrict__ x,
    const float* __restrict__ qw, const float* __restrict__ qb,
    const float* __restrict__ kw, const float* __restrict__ kb,
    const float* __restrict__ vw, const float* __restrict__ vb,
    __bf16* __restrict__ Qp, __bf16* __restrict__ Kp, __bf16* __restrict__ Vt)
{
    __shared__ float xt[CCH][65];   // [channel][token], padded
    const int b  = blockIdx.y;
    const int s0 = blockIdx.x * 64;
    const int tid = threadIdx.x;

    for (int i = tid; i < CCH * 64; i += 256) {
        int c = i >> 6, s = i & 63;
        xt[c][s] = x[((size_t)b * CCH + c) * STOK + s0 + s];
    }
    __syncthreads();

    const int tok = tid & 63;
    const int grp = tid >> 6;           // 4 groups x 16 out-channels
    const size_t sidx = (size_t)b * STOK + s0 + tok;

    for (int oc = grp * 16; oc < grp * 16 + 16; ++oc) {
        float aq = qb[oc], ak = kb[oc], av = vb[oc];
        #pragma unroll 8
        for (int c = 0; c < CCH; ++c) {
            float xv = xt[c][tok];
            aq += xv * qw[oc * CCH + c];
            ak += xv * kw[oc * CCH + c];
            av += xv * vw[oc * CCH + c];
        }
        Qp[sidx * CCH + oc] = (__bf16)(aq * QSCALE);
        Kp[sidx * CCH + oc] = (__bf16)ak;
        Vt[((size_t)b * CCH + oc) * STOK + s0 + tok] = (__bf16)av;
    }
}

// ---------------------------------------------------------------------------
// Kernel 2: flash attention, bf16 WMMA 16x16x32, fp32 accum.
// One wave owns 16 queries. K/V streamed in 64-key chunks, double-buffered
// in LDS via async global->LDS copies (ASYNCcnt). 16 WMMAs per iteration.
// ---------------------------------------------------------------------------
__global__ __launch_bounds__(256) void fa_kernel(
    const __bf16* __restrict__ Qp, const __bf16* __restrict__ Kp,
    const __bf16* __restrict__ Vt, float* __restrict__ Ob)
{
    // 144B padded rows: 16-lane b128 reads hit distinct banks
    __shared__ __align__(16) __bf16 ldsK[2][64][72];  // [buf][key][chan]
    __shared__ __align__(16) __bf16 ldsV[2][64][72];  // [buf][chan][key]
    __shared__ __align__(16) __bf16 ldsP[8][16][72];  // per-wave P tile [row][key]

    const int tid  = threadIdx.x;
    const int wave = tid >> 5;
    const int lane = tid & 31;
    const int half = lane >> 4;   // lane group 0..15 / 16..31
    const int l16  = lane & 15;
    const int b    = blockIdx.y;
    const int q0   = blockIdx.x * 128 + wave * 16;

    const __bf16* kbase = Kp + (size_t)b * STOK * CCH;
    const __bf16* vbase = Vt + (size_t)b * CCH * STOK;

    // --- async stage of one 64-key K/V chunk (8KB + 8KB) into buffer `buf` ---
    auto issue_tile = [&](int buf, int kc) {
        #pragma unroll
        for (int i = 0; i < 2; ++i) {
            int slot = tid + 256 * i;
            int row  = slot >> 3;          // K: key / V: channel
            int part = slot & 7;           // 8 x 16B per 128B row
            {   // K chunk: 64 keys x 64 ch
                unsigned int la = (unsigned int)(uintptr_t)&ldsK[buf][row][part * 8];
                unsigned long long ga = (unsigned long long)(uintptr_t)
                    (kbase + (size_t)(kc + row) * CCH + part * 8);
                asm volatile("global_load_async_to_lds_b128 %0, %1, off"
                             :: "v"(la), "v"(ga) : "memory");
            }
            {   // V chunk: 64 ch x 64 keys (Vt is channel-major)
                unsigned int la = (unsigned int)(uintptr_t)&ldsV[buf][row][part * 8];
                unsigned long long ga = (unsigned long long)(uintptr_t)
                    (vbase + (size_t)row * STOK + kc + part * 8);
                asm volatile("global_load_async_to_lds_b128 %0, %1, off"
                             :: "v"(la), "v"(ga) : "memory");
            }
        }
    };

    // --- Q A-fragments (16x32 each): lane l: row=l16,
    //     v[0..3]: K = 8*half+{0..7}, v[4..7]: K = 16+8*half+{0..7} ---
    const __bf16* qrow = Qp + ((size_t)(b * STOK + q0 + l16)) * CCH;
    v16bf aq[2];
    aq[0] = ld_frag8_8(qrow + 8 * half);
    aq[1] = ld_frag8_8(qrow + 32 + 8 * half);

    v8f o[4] = {};
    float run_max[8], run_sum[8];
    #pragma unroll
    for (int j = 0; j < 8; ++j) { run_max[j] = -1e30f; run_sum[j] = 0.0f; }

    issue_tile(0, 0);

    for (int kc = 0; kc < STOK; kc += 64) {
        const int buf = (kc >> 6) & 1;
        asm volatile("s_wait_asynccnt 0x0" ::: "memory");
        __syncthreads();                       // all waves' chunk data visible
        if (kc + 64 < STOK) issue_tile(buf ^ 1, kc + 64);

        // ---- scores: S[16x64] = Q(16x64) x K(chunk)^T, 4 tiles x 2 WMMAs ----
        v8f s[4];
        #pragma unroll
        for (int nt = 0; nt < 4; ++nt) {
            v16bf b0 = ld_frag8_8(&ldsK[buf][16 * nt + l16][     8 * half]);  // c 0..31
            v16bf b1 = ld_frag8_8(&ldsK[buf][16 * nt + l16][32 + 8 * half]);  // c 32..63
            v8f acc = {};
            acc = __builtin_amdgcn_wmma_f32_16x16x32_bf16(false, aq[0], false, b0, (short)0, acc, false, false);
            acc = __builtin_amdgcn_wmma_f32_16x16x32_bf16(false, aq[1], false, b1, (short)0, acc, false, false);
            s[nt] = acc;
        }

        // ---- online softmax (base-2; scale folded into Q) ----
        float alpha[8];
        #pragma unroll
        for (int j = 0; j < 8; ++j) {
            float m = fmaxf(fmaxf(s[0][j], s[1][j]), fmaxf(s[2][j], s[3][j]));
            m = row_max16(m);
            float nm = fmaxf(run_max[j], m);
            alpha[j] = __builtin_amdgcn_exp2f(run_max[j] - nm);
            run_max[j] = nm;
            float p0 = __builtin_amdgcn_exp2f(s[0][j] - nm);
            float p1 = __builtin_amdgcn_exp2f(s[1][j] - nm);
            float p2 = __builtin_amdgcn_exp2f(s[2][j] - nm);
            float p3 = __builtin_amdgcn_exp2f(s[3][j] - nm);
            run_sum[j] = run_sum[j] * alpha[j] + ((p0 + p1) + (p2 + p3));
            int r = j + 8 * half;
            ldsP[wave][r][l16]      = (__bf16)p0;
            ldsP[wave][r][l16 + 16] = (__bf16)p1;
            ldsP[wave][r][l16 + 32] = (__bf16)p2;
            ldsP[wave][r][l16 + 48] = (__bf16)p3;
        }
        asm volatile("s_wait_dscnt 0x0" ::: "memory");

        #pragma unroll
        for (int t = 0; t < 4; ++t)
            #pragma unroll
            for (int j = 0; j < 8; ++j)
                o[t][j] *= alpha[j];

        // ---- P back as two A-fragments (16x32 each, already bf16) ----
        v16bf ap0 = ld_frag8_8(&ldsP[wave][l16][     8 * half]);   // keys 0..31
        v16bf ap1 = ld_frag8_8(&ldsP[wave][l16][32 + 8 * half]);   // keys 32..63

        // ---- O(16x64) += P(16x64) x V(chunk), 4 tiles x 2 WMMAs ----
        #pragma unroll
        for (int t = 0; t < 4; ++t) {
            v16bf bvlo = ld_frag8_8(&ldsV[buf][16 * t + l16][     8 * half]); // keys 0..31
            v16bf bvhi = ld_frag8_8(&ldsV[buf][16 * t + l16][32 + 8 * half]); // keys 32..63
            o[t] = __builtin_amdgcn_wmma_f32_16x16x32_bf16(false, ap0, false, bvlo, (short)0, o[t], false, false);
            o[t] = __builtin_amdgcn_wmma_f32_16x16x32_bf16(false, ap1, false, bvhi, (short)0, o[t], false, false);
        }
    }

    // ---- final row-sum reduce, normalize, store O [b][q][c] fp32 ----
    float* obase = Ob + ((size_t)(b * STOK + q0)) * CCH;
    #pragma unroll
    for (int j = 0; j < 8; ++j) {
        float inv = 1.0f / row_sum16(run_sum[j]);
        int row = j + 8 * half;
        #pragma unroll
        for (int t = 0; t < 4; ++t)
            obase[(size_t)row * CCH + 16 * t + l16] = o[t][j] * inv;
    }
}

// ---------------------------------------------------------------------------
// Kernel 3: output projection + transpose back to [B,C,H,W]
// ---------------------------------------------------------------------------
__global__ __launch_bounds__(256) void oproj_kernel(
    const float* __restrict__ Ob, const float* __restrict__ ow,
    const float* __restrict__ obias, float* __restrict__ out)
{
    __shared__ float ot[64][65];   // [token][channel], padded
    const int b  = blockIdx.y;
    const int s0 = blockIdx.x * 64;
    const int tid = threadIdx.x;

    for (int i = tid; i < 64 * CCH; i += 256) {
        int s = i >> 6, k = i & 63;
        ot[s][k] = Ob[((size_t)(b * STOK + s0 + s)) * CCH + k];
    }
    __syncthreads();

    const int tok = tid & 63;
    const int grp = tid >> 6;
    for (int oc = grp * 16; oc < grp * 16 + 16; ++oc) {
        float acc = obias[oc];
        #pragma unroll 8
        for (int k = 0; k < CCH; ++k)
            acc += ot[tok][k] * ow[oc * CCH + k];
        out[((size_t)b * CCH + oc) * STOK + s0 + tok] = acc;
    }
}

// ---------------------------------------------------------------------------
extern "C" void kernel_launch(void* const* d_in, const int* in_sizes, int n_in,
                              void* d_out, int out_size, void* d_ws, size_t ws_size,
                              hipStream_t stream) {
    const float* x   = (const float*)d_in[0];
    const float* qw  = (const float*)d_in[1];
    const float* qb  = (const float*)d_in[2];
    const float* kw  = (const float*)d_in[3];
    const float* kb  = (const float*)d_in[4];
    const float* vw  = (const float*)d_in[5];
    const float* vb  = (const float*)d_in[6];
    const float* ow  = (const float*)d_in[7];
    const float* ob  = (const float*)d_in[8];
    float* out = (float*)d_out;

    // workspace: Q (8MB bf16) | K (8MB bf16) | Vt (8MB bf16) | O (16MB f32) = 40MB
    char* ws = (char*)d_ws;
    __bf16* Qp = (__bf16*)(ws);
    __bf16* Kp = (__bf16*)(ws + (size_t)8  * 1024 * 1024);
    __bf16* Vt = (__bf16*)(ws + (size_t)16 * 1024 * 1024);
    float*  Ob = (float*) (ws + (size_t)24 * 1024 * 1024);

    qkv_kernel  <<<dim3(STOK / 64,  BATCH), 256, 0, stream>>>(x, qw, qb, kw, kb, vw, vb, Qp, Kp, Vt);
    fa_kernel   <<<dim3(STOK / 128, BATCH), 256, 0, stream>>>(Qp, Kp, Vt, Ob);
    oproj_kernel<<<dim3(STOK / 64,  BATCH), 256, 0, stream>>>(Ob, ow, ob, out);
}
